// MLA_10488310137112
// MI455X (gfx1250) — compile-verified, matrix operand order checked by reading, not run on hardware
//
#include <hip/hip_runtime.h>
#include <hip/hip_bf16.h>

// ---------------------------------------------------------------------------
// CDNA5 (gfx1250) WMMA helpers
// ---------------------------------------------------------------------------
typedef __attribute__((ext_vector_type(16))) __bf16 v16bf;
typedef __attribute__((ext_vector_type(8)))  float  v8f;
typedef __attribute__((ext_vector_type(4)))  unsigned u32x4;
typedef __attribute__((ext_vector_type(8)))  int      i32x8;
typedef __attribute__((ext_vector_type(4)))  int      i32x4;

union Frag16 { unsigned u[8]; v16bf v; };

__device__ inline v8f vzero() { v8f z = {0.f,0.f,0.f,0.f,0.f,0.f,0.f,0.f}; return z; }

__device__ inline v8f wmma_bf16(const Frag16& a, const Frag16& b, v8f c) {
  return __builtin_amdgcn_wmma_f32_16x16x32_bf16(false, a.v, false, b.v,
                                                 (short)0, c, false, false);
}

// K index of dword j inside a 16-bit 16x32 A/B fragment (ISA 7.12.2):
// lanes 0-15: dwords 0-3 -> K 0..7,  dwords 4-7 -> K 16..23
// lanes 16-31: dwords 0-3 -> K 8..15, dwords 4-7 -> K 24..31
__device__ inline int kbase(int j, int half) {
  return (j < 4 ? 2 * j : 16 + 2 * (j - 4)) + 8 * half;
}

#if __has_builtin(__builtin_amdgcn_tensor_load_to_lds) && \
    __has_builtin(__builtin_amdgcn_s_wait_tensorcnt)
#define HAVE_TDM 1
#else
#define HAVE_TDM 0
#endif

// ---------------------------------------------------------------------------
// fp32 -> bf16 bulk convert
// ---------------------------------------------------------------------------
__global__ void f32_to_bf16_kernel(const float* __restrict__ src,
                                   __bf16* __restrict__ dst, int n) {
  int i = blockIdx.x * blockDim.x + threadIdx.x;
  if (i < n) dst[i] = (__bf16)src[i];
}

// ---------------------------------------------------------------------------
// bf16-in WMMA GEMM: C[MxN] = A[MxK] @ B[KxN], row-major, OutT in {__bf16,float}.
// Requires M%128==0, N%128==0, K%32==0 (true for every GEMM here).
// Block 256 threads (8 waves), tile 128x128, K-step 32, wave grid 4x2 (2x4 WMMA).
// A tile staged by the Tensor Data Mover when available (D# pad feature emits the
// 64B row + 8B pad = LDST stride); B tile staged transposed by all waves.
// ---------------------------------------------------------------------------
#define TBM 128
#define TBN 128
#define TBK 32
#define LDST 36   // padded bf16 row stride: 72B rows -> 8B-aligned, 18-dword stride

template <typename OutT>
__global__ __launch_bounds__(256)
void gemm_bf16io_kernel(const __bf16* __restrict__ A, const __bf16* __restrict__ B,
                        OutT* __restrict__ C, int M, int N, int K) {
  __shared__ __bf16 As[TBM * LDST];      // [row][k]  k contiguous
  __shared__ __bf16 Bs[TBN * LDST];      // [col][k]  transposed: k contiguous
  const int tid  = threadIdx.x;
  const int wave = tid >> 5, lane = tid & 31;
  const int r = lane & 15, half = lane >> 4;
  const int waveRow = wave >> 1, waveCol = wave & 1;   // 4 x 2 waves
  const int rowBase = blockIdx.y * TBM;
  const int colBase = blockIdx.x * TBN;

  v8f acc[2][4];
  for (int i = 0; i < 2; ++i)
    for (int j = 0; j < 4; ++j) acc[i][j] = vzero();

  for (int k0 = 0; k0 < K; k0 += TBK) {
#if HAVE_TDM
    // ---- A tile via Tensor Data Mover: 128x32 bf16, row stride K, LDS-padded ----
    if (wave == 0) {
      const unsigned lds_a = (unsigned)(size_t)(__bf16*)As;  // LDS byte offset
      const unsigned long long ga =
          (unsigned long long)(const void*)&A[(size_t)rowBase * K + k0];
      u32x4 g0;
      g0[0] = 1u;                                   // count=1 valid descriptor
      g0[1] = lds_a;                                // lds_addr
      g0[2] = (unsigned)(ga & 0xffffffffu);         // global_addr[31:0]
      g0[3] = (unsigned)((ga >> 32) & 0x1ffffffu)   // global_addr[56:32]
            | (2u << 30);                           // type=2 ("image")
      i32x8 g1;
      g1[0] = (int)((1u << 16)                      // data_size = 2 bytes
                  | (1u << 20)                      // pad_enable
                  | (3u << 22)                      // pad_interval: 16 dwords (64B row)
                  | (1u << 25));                    // pad_amount: 2 dwords (8B)
      g1[1] = (int)(((unsigned)K & 0xffffu) << 16); // tensor_dim0[15:0]
      g1[2] = (int)(((unsigned)K >> 16)             // tensor_dim0[31:16]
                  | (((unsigned)M & 0xffffu) << 16)); // tensor_dim1[15:0]
      g1[3] = (int)(((unsigned)M >> 16)             // tensor_dim1[31:16]
                  | ((unsigned)TBK << 16));         // tile_dim0 = 32
      g1[4] = TBM;                                  // tile_dim1 = 128 (tile_dim2=0)
      g1[5] = K;                                    // tensor_dim0_stride[31:0]
      g1[6] = 0; g1[7] = 0;
      i32x4 gz = {0, 0, 0, 0};
#if __clang_major__ >= 23
      i32x8 gz8 = {0, 0, 0, 0, 0, 0, 0, 0};
      __builtin_amdgcn_tensor_load_to_lds(g0, g1, gz, gz, gz8, 0);
#else
      __builtin_amdgcn_tensor_load_to_lds(g0, g1, gz, gz, 0);
#endif
      __builtin_amdgcn_s_wait_tensorcnt(0);
    }
#else
    // ---- manual A tile staging: straight b64 copies ----
    {
      const int quad = tid & 7;        // 8 x (4 bf16) per 32-wide row
      const int rr   = tid >> 3;       // 32 rows per pass
      for (int p = 0; p < 4; ++p) {
        const int row = p * 32 + rr;
        const uint2 d = *(const uint2*)&A[(size_t)(rowBase + row) * K + k0 + quad * 4];
        *(uint2*)&As[row * LDST + quad * 4] = d;
      }
    }
#endif
    // ---- B tile 32x128 staged transposed to Bs[col][k] (bf16 K-pairs = dwords) ----
    {
      const int c4  = (tid & 31) * 4;
      const int kr0 = tid >> 5;        // 8 k-rows per pass
      for (int p = 0; p < 4; ++p) {
        const int kr = p * 8 + kr0;
        union { uint2 d; __bf16 h[4]; } u;
        u.d = *(const uint2*)&B[(size_t)(k0 + kr) * N + colBase + c4];
        Bs[(c4 + 0) * LDST + kr] = u.h[0];
        Bs[(c4 + 1) * LDST + kr] = u.h[1];
        Bs[(c4 + 2) * LDST + kr] = u.h[2];
        Bs[(c4 + 3) * LDST + kr] = u.h[3];
      }
    }
    // Prefetch next K-tile into cache while this one computes.
    if (k0 + TBK < K) {
      __builtin_prefetch(&A[(size_t)(rowBase + (tid & 127)) * K + k0 + TBK], 0, 3);
      __builtin_prefetch(&B[(size_t)(k0 + TBK + (tid >> 3)) * N + colBase + (tid & 7) * 16], 0, 3);
    }
    __syncthreads();

    Frag16 af[2], bf[4];
    for (int mt = 0; mt < 2; ++mt) {
      const int row = waveRow * 32 + mt * 16 + r;
      for (int j = 0; j < 8; ++j)
        af[mt].u[j] = *(const unsigned*)&As[row * LDST + kbase(j, half)];
    }
    for (int nt = 0; nt < 4; ++nt) {
      const int col = waveCol * 64 + nt * 16 + r;
      for (int j = 0; j < 8; ++j)
        bf[nt].u[j] = *(const unsigned*)&Bs[col * LDST + kbase(j, half)];
    }
    for (int mt = 0; mt < 2; ++mt)
      for (int nt = 0; nt < 4; ++nt)
        acc[mt][nt] = wmma_bf16(af[mt], bf[nt], acc[mt][nt]);
    __syncthreads();
  }

  // C/D layout: lane n = lane&15, VGPR i -> row i + 8*(lane>=16)
  for (int mt = 0; mt < 2; ++mt)
    for (int nt = 0; nt < 4; ++nt)
      for (int i = 0; i < 8; ++i) {
        const int row = rowBase + waveRow * 32 + mt * 16 + i + 8 * half;
        const int col = colBase + waveCol * 64 + nt * 16 + r;
        C[(size_t)row * N + col] = (OutT)acc[mt][nt][i];
      }
}

// ---------------------------------------------------------------------------
// Pack: out[b,s,h,0:64] = nope*scale ; out[b,s,h,64:128] = rope(ropein)*scale
// half = 32. One thread per (b,s,h,d<32) handles 4 outputs.
// ---------------------------------------------------------------------------
#define NHD   16
#define DH    128

__global__ void pack_rope_kernel(const __bf16* __restrict__ nope,
                                 const __bf16* __restrict__ ropein,
                                 __bf16* __restrict__ outp,
                                 int B, int S, int H, float scale) {
  const int total = B * S * H * 32;
  int idx = blockIdx.x * blockDim.x + threadIdx.x;
  if (idx >= total) return;
  const int d = idx & 31;
  const int h = (idx >> 5) & (H - 1);
  const int s = (idx / (32 * H)) % S;
  const int b = idx / (32 * H * S);
  const size_t ibase = ((size_t)(b * S + s) * H + h) * 64;
  const size_t obase = ((size_t)(b * S + s) * H + h) * 128;
  outp[obase + d]      = (__bf16)((float)nope[ibase + d]      * scale);
  outp[obase + d + 32] = (__bf16)((float)nope[ibase + d + 32] * scale);
  const float inv = __powf(10000.f, -(float)d / 32.f);
  const float ang = (float)s * inv;
  const float c = __cosf(ang), sn = __sinf(ang);
  const float x1 = (float)ropein[ibase + d], x2 = (float)ropein[ibase + d + 32];
  outp[obase + 64 + d]      = (__bf16)((x1 * c - x2 * sn) * scale);
  outp[obase + 64 + d + 32] = (__bf16)((x2 * c + x1 * sn) * scale);
}

// ---------------------------------------------------------------------------
// V transpose: vt[b,h,f,s] = v[b,s,h,f]  (so key-pairs are contiguous dwords)
// ---------------------------------------------------------------------------
__global__ void transpose_v_kernel(const __bf16* __restrict__ v,
                                   __bf16* __restrict__ vt, int B, int S, int H) {
  const int total = B * S * H * DH;
  int idx = blockIdx.x * blockDim.x + threadIdx.x;
  if (idx >= total) return;
  const int f = idx & (DH - 1);
  const int h = (idx >> 7) & (H - 1);
  const int s = (idx / (DH * H)) % S;
  const int b = idx / (DH * H * S);
  vt[((size_t)(b * H + h) * DH + f) * S + s] = v[idx];
}

// ---------------------------------------------------------------------------
// Causal flash attention, all-bf16 operands, fp32 accumulation.
// q16/k16: [b,s,h,128] (scale folded into q16), vt16: [b,h,128,S].
// Block 128 threads (4 waves); wave = 16 query rows; 32 keys per iteration.
// ---------------------------------------------------------------------------
#define PSTR  36   // bf16 stride of P staging rows

__global__ __launch_bounds__(128)
void mla_attn_kernel(const __bf16* __restrict__ q16, const __bf16* __restrict__ k16,
                     const __bf16* __restrict__ vt16, __bf16* __restrict__ outp, int S) {
  __shared__ __bf16 Ps[4][16 * PSTR];    // per-wave P staging (C-layout -> A-layout)
  const int tid  = threadIdx.x;
  const int wave = tid >> 5, lane = tid & 31;
  const int r = lane & 15, half = lane >> 4;
  const int h = blockIdx.y, b = blockIdx.z;
  const int q0 = blockIdx.x * 64 + wave * 16;

  // Q fragments: single base per lane + immediate offsets.
  Frag16 qf[4];
  {
    const __bf16* qptr = q16 + ((size_t)(b * S + q0 + r) * NHD + h) * DH;
    for (int c = 0; c < 4; ++c)
      for (int j = 0; j < 8; ++j)
        qf[c].u[j] = *(const unsigned*)(qptr + c * 32 + kbase(j, half));
  }

  float mrow[8], lrow[8];
  v8f acc[8];
  for (int i = 0; i < 8; ++i) { mrow[i] = -1e30f; lrow[i] = 0.f; acc[i] = vzero(); }

  const int kend = q0 + 16;                   // causal bound (<= S always)
  for (int ks = 0; ks < kend; ks += 32) {
    const __bf16* vbase = vt16 + ((size_t)(b * NHD + h) * DH + r) * S + ks;

    // Prefetch next key chunk (K rows + V columns) while computing this one.
    if (ks + 32 < kend) {
      const __bf16* knext = k16 + ((size_t)(b * S + ks + 32 + lane) * NHD + h) * DH;
      __builtin_prefetch(knext, 0, 3);
      __builtin_prefetch(knext + 64, 0, 3);
      __builtin_prefetch(vt16 + ((size_t)(b * NHD + h) * DH + lane * 4) * S + ks + 32, 0, 3);
    }

    // ---- scores: two 16-key tiles, each a chain of 4 WMMAs over d=128 ----
    v8f sc[2];
    for (int t = 0; t < 2; ++t) {
      sc[t] = vzero();
      const __bf16* kptr = k16 + ((size_t)(b * S + ks + t * 16 + r) * NHD + h) * DH;
      for (int c = 0; c < 4; ++c) {
        Frag16 kf;
        for (int j = 0; j < 8; ++j)
          kf.u[j] = *(const unsigned*)(kptr + c * 32 + kbase(j, half));
        sc[t] = wmma_bf16(qf[c], kf, sc[t]);
      }
    }

    // ---- causal mask + online softmax (16-lane row reductions) ----
    float rm[8];
    for (int i = 0; i < 8; ++i) {
      const int qrow = q0 + i + 8 * half;
      if (ks + r      > qrow) sc[0][i] = -1e30f;
      if (ks + 16 + r > qrow) sc[1][i] = -1e30f;
      float x = fmaxf(sc[0][i], sc[1][i]);
      for (int m = 1; m < 16; m <<= 1) x = fmaxf(x, __shfl_xor(x, m, 16));
      rm[i] = x;
    }
    float alpha[8];
    for (int i = 0; i < 8; ++i) {
      const float mn = fmaxf(mrow[i], rm[i]);
      alpha[i] = __expf(mrow[i] - mn);
      sc[0][i] = __expf(sc[0][i] - mn);
      sc[1][i] = __expf(sc[1][i] - mn);
      float rs = sc[0][i] + sc[1][i];
      for (int m = 1; m < 16; m <<= 1) rs += __shfl_xor(rs, m, 16);
      lrow[i] = lrow[i] * alpha[i] + rs;
      mrow[i] = mn;
    }
    for (int nt = 0; nt < 8; ++nt)
      for (int i = 0; i < 8; ++i) acc[nt][i] *= alpha[i];

    // ---- P: C-layout -> bf16 A-layout via per-wave LDS (wave-local sync) ----
    for (int t = 0; t < 2; ++t)
      for (int i = 0; i < 8; ++i)
        Ps[wave][(i + 8 * half) * PSTR + t * 16 + r] = (__bf16)sc[t][i];
    asm volatile("s_wait_dscnt 0" ::: "memory");
    Frag16 pf;
    for (int j = 0; j < 8; ++j)
      pf.u[j] = *(const unsigned*)&Ps[wave][r * PSTR + kbase(j, half)];

    // ---- acc += P(16x32) @ V(32x128): key-pairs contiguous in vt16 ----
    for (int nt = 0; nt < 8; ++nt) {
      Frag16 vf;
      for (int j = 0; j < 8; ++j)
        vf.u[j] = *(const unsigned*)(vbase + (size_t)nt * 16 * S + kbase(j, half));
      acc[nt] = wmma_bf16(pf, vf, acc[nt]);
    }
  }

  for (int nt = 0; nt < 8; ++nt)
    for (int i = 0; i < 8; ++i) {
      const int q = q0 + i + 8 * half;
      outp[((size_t)(b * S + q) * NHD + h) * DH + nt * 16 + r] =
          (__bf16)(acc[nt][i] / lrow[i]);
    }
}

// ---------------------------------------------------------------------------
// Launch
// ---------------------------------------------------------------------------
extern "C" void kernel_launch(void* const* d_in, const int* in_sizes, int n_in,
                              void* d_out, int out_size, void* d_ws, size_t ws_size,
                              hipStream_t stream) {
  const float* xf = (const float*)d_in[0];
  float* out = (float*)d_out;

  const int B = 2, S = 2048, HD = 2048, CD = 512, NH = 16;
  const int BS = B * S;                         // 4096 rows

  // ---- bf16 workspace layout (element offsets; all 8B-aligned) ----
  __bf16* ws = (__bf16*)d_ws;
  size_t off = 0;
  auto alloc = [&](size_t n) { __bf16* p = ws + off; off += n; return p; };
  __bf16* x16    = alloc((size_t)BS * HD);       // 8M  (reused as attn16 later)
  __bf16* w16[8];
  const int wsz[8] = {HD * CD, HD * CD, CD * 1024, CD * 1024, CD * 2048,
                      HD * 1024, CD * 1024, HD * HD};
  // weights: W_dkv W_dq W_uq W_uk W_uv W_kr W_qr W_o
  for (int i = 0; i < 8; ++i) w16[i] = alloc((size_t)wsz[i]);
  __bf16* c_kv16 = alloc((size_t)BS * CD);
  __bf16* c_q16  = alloc((size_t)BS * CD);
  __bf16* q_c16  = alloc((size_t)BS * 1024);
  __bf16* k_c16  = alloc((size_t)BS * 1024);
  __bf16* v16    = alloc((size_t)BS * 2048);
  __bf16* k_r16  = alloc((size_t)BS * 1024);
  __bf16* q_r16  = alloc((size_t)BS * 1024);
  __bf16* q16    = alloc((size_t)BS * 2048);
  __bf16* k16    = alloc((size_t)BS * 2048);
  __bf16* vt16   = alloc((size_t)BS * 2048);
  __bf16* attn16 = x16;                          // x dead after projections

  auto cvt = [&](const void* src, __bf16* dst, int n) {
    f32_to_bf16_kernel<<<(n + 255) / 256, 256, 0, stream>>>((const float*)src, dst, n);
  };
  cvt(xf, x16, BS * HD);
  for (int i = 0; i < 8; ++i) cvt(d_in[1 + i], w16[i], wsz[i]);

  auto gemm16 = [&](const __bf16* A, const __bf16* Bm, __bf16* C, int M, int N, int K) {
    dim3 g(N / TBN, M / TBM);
    gemm_bf16io_kernel<__bf16><<<g, 256, 0, stream>>>(A, Bm, C, M, N, K);
  };

  gemm16(x16,    w16[0], c_kv16, BS, CD,   HD);   // c_kv
  gemm16(x16,    w16[1], c_q16,  BS, CD,   HD);   // c_q
  gemm16(c_q16,  w16[2], q_c16,  BS, 1024, CD);   // q_c
  gemm16(c_kv16, w16[3], k_c16,  BS, 1024, CD);   // k_c
  gemm16(c_kv16, w16[4], v16,    BS, 2048, CD);   // v
  gemm16(x16,    w16[5], k_r16,  BS, 1024, HD);   // k_r (pre-rope)
  gemm16(c_q16,  w16[6], q_r16,  BS, 1024, CD);   // q_r (pre-rope)

  const float scale = 0.08838834764831845f;       // 1/sqrt(128)
  const int npack = B * S * NH * 32;
  pack_rope_kernel<<<(npack + 255) / 256, 256, 0, stream>>>(q_c16, q_r16, q16,
                                                            B, S, NH, scale);
  pack_rope_kernel<<<(npack + 255) / 256, 256, 0, stream>>>(k_c16, k_r16, k16,
                                                            B, S, NH, 1.0f);
  const int nv = B * S * NH * DH;
  transpose_v_kernel<<<(nv + 255) / 256, 256, 0, stream>>>(v16, vt16, B, S, NH);

  mla_attn_kernel<<<dim3(S / 64, NH, B), 128, 0, stream>>>(q16, k16, vt16, attn16, S);

  // out = attn @ W_o  (fp32 output)
  dim3 g(HD / TBN, BS / TBM);
  gemm_bf16io_kernel<float><<<g, 256, 0, stream>>>(attn16, w16[7], out, BS, HD, HD);
}
